// AdditiveAttention_20289425507159
// MI455X (gfx1250) — matrix operand order repaired and checked by reading, not compiled, not run on hardware
//
#include <hip/hip_runtime.h>
#include <hip/hip_bf16.h>

typedef __attribute__((ext_vector_type(16))) _Float16 v16h;
typedef __attribute__((ext_vector_type(4)))  _Float16 v4h;
typedef __attribute__((ext_vector_type(8)))  float    v8f;

#define B_  4
#define Q_  1024
#define K_  1024
#define QS_ 256
#define H_  64
#define V_  256

#if __has_builtin(__builtin_amdgcn_tanhf)
#define TANHF(x) __builtin_amdgcn_tanhf(x)
#else
#define TANHF(x) tanhf(x)
#endif

// ---------------------------------------------------------------------------
// CDNA5 async global->LDS copy (ASYNCcnt path), portable via inline asm.
// LDS byte address = low 32 bits of the generic pointer (LDS aperture puts
// the DS offset in addr[31:0]).
// ---------------------------------------------------------------------------
__device__ __forceinline__ void async_ld_b128(void* lds_dst, const void* gsrc) {
    unsigned l = (unsigned)(unsigned long long)lds_dst;
    asm volatile("global_load_async_to_lds_b128 %0, %1, off"
                 :: "v"(l), "v"(gsrc) : "memory");
}
__device__ __forceinline__ void wait_async0() {
    asm volatile("s_wait_asynccnt 0" ::: "memory");
}

// ---------------------------------------------------------------------------
// Kernel 0: values f32 -> f16 (feeds async b128 staging in av_kernel)
// ---------------------------------------------------------------------------
__global__ void cvt_values_kernel(const float* __restrict__ x,
                                  _Float16* __restrict__ y) {
    int i = blockIdx.x * blockDim.x + threadIdx.x;   // over float4 groups
    float4 v = ((const float4*)x)[i];
    v4h o = { (_Float16)v.x, (_Float16)v.y, (_Float16)v.z, (_Float16)v.w };
    ((v4h*)y)[i] = o;
}

// ---------------------------------------------------------------------------
// Kernel 1: Y[M,64] = X[M,256] * W[64,256]^T   (WMMA f16, f32 accumulate)
// block = 128 threads (4 waves); each wave owns one 16-wide N tile.
// ---------------------------------------------------------------------------
__global__ void proj_kernel(const float* __restrict__ X,
                            const float* __restrict__ W,
                            float* __restrict__ Y) {
    __shared__ _Float16 Alds[16 * QS_];   // 8 KB
    __shared__ _Float16 Wlds[H_ * QS_];   // 32 KB
    const int tid   = threadIdx.x;
    const int wave  = tid >> 5;           // 0..3 -> N tile
    const int lane  = tid & 31;
    const int l16   = lane & 15;
    const int lhi   = lane >> 4;          // 0 or 1
    const int mbase = blockIdx.x * 16;

    for (int i = tid; i < 16 * QS_; i += 128) {
        int r = i >> 8, c = i & 255;
        Alds[i] = (_Float16)X[(size_t)(mbase + r) * QS_ + c];
    }
    for (int i = tid; i < H_ * QS_; i += 128)
        Wlds[i] = (_Float16)W[i];
    __syncthreads();

    v8f acc = {};
    for (int ks = 0; ks < QS_; ks += 32) {
        v16h a, b;
#pragma unroll
        for (int v = 0; v < 8; ++v) {
            int ka = ((v >> 2) * 16) + (lhi * 8) + ((v & 3) * 2);
            a[2 * v + 0] = Alds[l16 * QS_ + ks + ka + 0];
            a[2 * v + 1] = Alds[l16 * QS_ + ks + ka + 1];
            int kb = ((v >> 2) * 16) + (lhi * 8) + ((v & 3) * 2);
            int n  = wave * 16 + l16;
            b[2 * v + 0] = Wlds[n * QS_ + ks + kb + 0];   // B[k,n] = W[n,k]
            b[2 * v + 1] = Wlds[n * QS_ + ks + kb + 1];
        }
        acc = __builtin_amdgcn_wmma_f32_16x16x32_f16(
            false, a, false, b, (short)0, acc, false, false);
    }
#pragma unroll
    for (int r = 0; r < 8; ++r) {
        int m = mbase + r + lhi * 8;
        int n = wave * 16 + l16;
        Y[(size_t)m * H_ + n] = acc[r];
    }
}

// ---------------------------------------------------------------------------
// Kernel 2: scores + masked softmax -> attn (f16).
// One 256-thread workgroup per (b,q) row; each thread owns 4 key positions.
// ---------------------------------------------------------------------------
__global__ void score_softmax_kernel(const float* __restrict__ qproj,
                                     const float* __restrict__ kproj,
                                     const float* __restrict__ w_v,
                                     const int*   __restrict__ valid_lens,
                                     _Float16*    __restrict__ attn) {
    __shared__ float qp[H_];
    __shared__ float wv[H_];
    __shared__ float red[256];

    const int bq  = blockIdx.x;        // b*Q + q
    const int b   = bq >> 10;
    const int tid = threadIdx.x;

    if (tid < H_) {
        qp[tid] = qproj[(size_t)bq * H_ + tid];
        wv[tid] = w_v[tid];
    }
    __syncthreads();

    const int    vlen = valid_lens[b];
    const float* kp   = kproj + (size_t)b * K_ * H_;

    float s[4];
    float lmax = -1.0e30f;
#pragma unroll
    for (int j = 0; j < 4; ++j) {
        int k = tid + j * 256;
        const float4* kr = (const float4*)(kp + (size_t)k * H_);
        if (j < 3)  // CDNA5 global_prefetch_b8 for next key row
            __builtin_prefetch(kp + (size_t)(k + 256) * H_, 0, 1);
        float acc = 0.f;
#pragma unroll
        for (int h4 = 0; h4 < H_ / 4; ++h4) {
            float4 kv = kr[h4];
            acc += wv[4 * h4 + 0] * TANHF(qp[4 * h4 + 0] + kv.x);
            acc += wv[4 * h4 + 1] * TANHF(qp[4 * h4 + 1] + kv.y);
            acc += wv[4 * h4 + 2] * TANHF(qp[4 * h4 + 2] + kv.z);
            acc += wv[4 * h4 + 3] * TANHF(qp[4 * h4 + 3] + kv.w);
        }
        s[j] = (k < vlen) ? acc : -1.0e6f;
        lmax = fmaxf(lmax, s[j]);
    }

    red[tid] = lmax;
    __syncthreads();
    for (int off = 128; off >= 1; off >>= 1) {
        if (tid < off) red[tid] = fmaxf(red[tid], red[tid + off]);
        __syncthreads();
    }
    const float gmax = red[0];
    __syncthreads();

    float e[4];
    float lsum = 0.f;
#pragma unroll
    for (int j = 0; j < 4; ++j) {
        e[j] = __expf(s[j] - gmax);
        lsum += e[j];
    }
    red[tid] = lsum;
    __syncthreads();
    for (int off = 128; off >= 1; off >>= 1) {
        if (tid < off) red[tid] += red[tid + off];
        __syncthreads();
    }
    const float inv = 1.0f / red[0];
#pragma unroll
    for (int j = 0; j < 4; ++j) {
        int k = tid + j * 256;
        attn[(size_t)bq * K_ + k] = (_Float16)(e[j] * inv);
    }
}

// ---------------------------------------------------------------------------
// Kernel 3: out[b,Q,V] = attn[b,Q,K] (f16) @ values16[b,K,V] (f16).
// block = 256 (8 waves); workgroup computes a 16(M) x 128(N) tile.
// Tiles staged LDS-direct with global_load_async_to_lds_b128 (ASYNCcnt).
// ---------------------------------------------------------------------------
__global__ void av_kernel(const _Float16* __restrict__ attn,
                          const _Float16* __restrict__ vals16,
                          float*          __restrict__ out) {
    __shared__ _Float16 Aat[16 * 32];    // 1 KB
    __shared__ _Float16 Bv[32 * 128];    // 8 KB
    const int b     = blockIdx.z;
    const int mbase = blockIdx.x * 16;
    const int nbase = blockIdx.y * 128;
    const int tid   = threadIdx.x;
    const int wave  = tid >> 5;          // 0..7 -> 16-wide N subtile
    const int lane  = tid & 31;
    const int l16   = lane & 15;
    const int lhi   = lane >> 4;

    v8f acc = {};
    for (int ks = 0; ks < K_; ks += 32) {
        __syncthreads();   // previous iteration's fragment reads done
        // attn tile: 16 rows x 32 halfs = 64 x 16B transfers
        if (tid < 64) {
            int r = tid >> 2, sgm = tid & 3;
            async_ld_b128(&Aat[r * 32 + sgm * 8],
                          &attn[((size_t)(b * Q_ + mbase + r)) * K_ + ks + sgm * 8]);
        }
        // values tile: 32 rows x 128 halfs = 512 x 16B transfers
#pragma unroll
        for (int i = tid; i < 512; i += 256) {
            int r = i >> 4, sgm = i & 15;
            async_ld_b128(&Bv[r * 128 + sgm * 8],
                          &vals16[((size_t)(b * K_ + ks + r)) * V_ + nbase + sgm * 8]);
        }
        wait_async0();
        __syncthreads();

        v16h a, bb;
#pragma unroll
        for (int v = 0; v < 8; ++v) {
            int ka = ((v >> 2) * 16) + (lhi * 8) + ((v & 3) * 2);
            a[2 * v + 0] = Aat[l16 * 32 + ka + 0];
            a[2 * v + 1] = Aat[l16 * 32 + ka + 1];
            int kb = ((v >> 2) * 16) + (lhi * 8) + ((v & 3) * 2);
            int n  = wave * 16 + l16;
            bb[2 * v + 0] = Bv[(kb + 0) * 128 + n];
            bb[2 * v + 1] = Bv[(kb + 1) * 128 + n];
        }
        acc = __builtin_amdgcn_wmma_f32_16x16x32_f16(
            false, a, false, bb, (short)0, acc, false, false);
    }
#pragma unroll
    for (int r = 0; r < 8; ++r) {
        int m = mbase + r + lhi * 8;
        int n = nbase + wave * 16 + l16;
        out[((size_t)(b * Q_ + m)) * V_ + n] = acc[r];
    }
}

// ---------------------------------------------------------------------------
extern "C" void kernel_launch(void* const* d_in, const int* in_sizes, int n_in,
                              void* d_out, int out_size, void* d_ws, size_t ws_size,
                              hipStream_t stream) {
    const float* queries    = (const float*)d_in[0];
    const float* keys       = (const float*)d_in[1];
    const float* values     = (const float*)d_in[2];
    const int*   valid_lens = (const int*)  d_in[3];
    const float* W_q        = (const float*)d_in[4];
    const float* W_k        = (const float*)d_in[5];
    const float* w_v        = (const float*)d_in[6];
    float*       out        = (float*)d_out;

    char* ws = (char*)d_ws;
    float*     qproj  = (float*)ws;                              // 1 MB
    float*     kproj  = (float*)(ws + ((size_t)1 << 20));        // 1 MB
    _Float16*  attn   = (_Float16*)(ws + ((size_t)2 << 20));     // 8 MB
    _Float16*  vals16 = (_Float16*)(ws + ((size_t)10 << 20));    // 2 MB

    // values f32 -> f16 (1M elements, processed as float4 groups)
    cvt_values_kernel<<<dim3((B_ * K_ * V_) / 4 / 256), 256, 0, stream>>>(values, vals16);

    // projections: M = B*Q = B*K = 4096 rows -> 256 tiles of 16
    proj_kernel<<<dim3(256), 128, 0, stream>>>(queries, W_q, qproj);
    proj_kernel<<<dim3(256), 128, 0, stream>>>(keys,    W_k, kproj);

    // scores + masked softmax: one workgroup per (b,q)
    score_softmax_kernel<<<dim3(B_ * Q_), 256, 0, stream>>>(
        qproj, kproj, w_v, valid_lens, attn);

    // attn @ values (async-LDS staged WMMA)
    av_kernel<<<dim3(Q_ / 16, V_ / 128, B_), 256, 0, stream>>>(attn, vals16, out);
}
